// T5TextConditionEncoder_17849884082438
// MI455X (gfx1250) — matrix operand order repaired and checked by reading, not compiled, not run on hardware
//
#include <hip/hip_runtime.h>
#include <hip/hip_bf16.h>

typedef __attribute__((ext_vector_type(8)))  float  v8f;
typedef __attribute__((ext_vector_type(16))) __bf16 v16bf;

// round-half-up bf16: add 0x8000 then take high 16 bits.
__device__ __forceinline__ unsigned pack_bf16(float a, float b) {
  unsigned ua = __float_as_uint(a) + 0x8000u;
  unsigned ub = __float_as_uint(b) + 0x8000u;
  // result = (hi16(ub) << 16) | hi16(ua)  -- one v_perm_b32
  return __builtin_amdgcn_perm(ub, ua, 0x07060302u);
}
__device__ __forceinline__ unsigned short bf16o(float f) {
  return (unsigned short)((__float_as_uint(f) + 0x8000u) >> 16);
}

union FragAB { v16bf v; unsigned u[8]; };

#define TM 128
#define TN 128
#define TK 32

// C[m,n] = sum_k A[m,k] * B[k,n]   (or B[n,k] when TRANSB)
// EPI: 0=none, 1=relu, 2=+residual(EA), 3=+bias[h]+mask[b] (attention scores)
// Assumes M%128==0, K%32==0 always; GUARD enables N-dimension bounds checks.
template<int EPI, bool TRANSB, bool GUARD>
__global__ __launch_bounds__(256)
void gemm_wmma(const float* __restrict__ A, int lda, long long Ab, long long Ah,
               const float* __restrict__ B, int ldb, long long Bb, long long Bh,
               float* __restrict__ C, int ldc, long long Cb, long long Ch,
               int M, int N, int K, int HH,
               const float* __restrict__ EA, const int* __restrict__ EB, int S) {
  __shared__ unsigned short As[TM][TK + 8];   // [m][k]
  __shared__ unsigned short Bs[TN][TK + 8];   // [n][k] (k-major for fragment reads)
  const int z = blockIdx.z;
  const int bidx = z / HH, hidx = z % HH;
  A += (long long)bidx * Ab + (long long)hidx * Ah;
  B += (long long)bidx * Bb + (long long)hidx * Bh;
  C += (long long)bidx * Cb + (long long)hidx * Ch;
  const int m0 = blockIdx.y * TM;
  const int n0 = blockIdx.x * TN;
  const int tid = threadIdx.x;
  const int lane = tid & 31;
  const int wave = tid >> 5;
  const int wm = (wave >> 2) * 64;   // 2 wave-rows of M
  const int wn = (wave & 3) * 32;    // 4 wave-cols of N

  // loop-invariant staging coordinates
  const int a_cq  = tid & 7;    // quad-of-k (A / transposed-B)
  const int a_r0  = tid >> 3;   // row base   (+32*i)
  const int bn_nq = tid & 31;   // quad-of-n  (normal B)
  const int bn_kt = tid >> 5;   // k base     (+8*i)

  float4 ra[4], rb[4];          // software-pipeline staging registers

  auto globalLoad = [&](int k0) {
#pragma unroll
    for (int i = 0; i < 4; ++i) {
      int r = 32 * i + a_r0;
      ra[i] = *(const float4*)&A[(long long)(m0 + r) * lda + k0 + 4 * a_cq];
    }
    if (TRANSB) {
#pragma unroll
      for (int i = 0; i < 4; ++i) {
        int r = 32 * i + a_r0;
        rb[i] = (!GUARD || (n0 + r) < N)
                    ? *(const float4*)&B[(long long)(n0 + r) * ldb + k0 + 4 * a_cq]
                    : make_float4(0.f, 0.f, 0.f, 0.f);
      }
    } else {
#pragma unroll
      for (int i = 0; i < 4; ++i) {
        int kk = 8 * i + bn_kt;
        rb[i] = (!GUARD || (n0 + 4 * bn_nq + 3) < N)
                    ? *(const float4*)&B[(long long)(k0 + kk) * ldb + n0 + 4 * bn_nq]
                    : make_float4(0.f, 0.f, 0.f, 0.f);
      }
    }
  };

  auto ldsStore = [&]() {
#pragma unroll
    for (int i = 0; i < 4; ++i) {
      int r = 32 * i + a_r0;
      *(uint2*)&As[r][4 * a_cq] =
          make_uint2(pack_bf16(ra[i].x, ra[i].y), pack_bf16(ra[i].z, ra[i].w));
    }
    if (TRANSB) {
#pragma unroll
      for (int i = 0; i < 4; ++i) {
        int r = 32 * i + a_r0;
        *(uint2*)&Bs[r][4 * a_cq] =
            make_uint2(pack_bf16(rb[i].x, rb[i].y), pack_bf16(rb[i].z, rb[i].w));
      }
    } else {
#pragma unroll
      for (int i = 0; i < 4; ++i) {
        int kk = 8 * i + bn_kt;
        int nb = 4 * bn_nq;
        Bs[nb + 0][kk] = bf16o(rb[i].x);
        Bs[nb + 1][kk] = bf16o(rb[i].y);
        Bs[nb + 2][kk] = bf16o(rb[i].z);
        Bs[nb + 3][kk] = bf16o(rb[i].w);
      }
    }
  };

  v8f acc[4][2] = {};

  globalLoad(0);
  ldsStore();
  __syncthreads();

  for (int k0 = 0; k0 < K; k0 += TK) {
    const bool more = (k0 + TK) < K;
    if (more) globalLoad(k0 + TK);   // vmem in flight during the math below

    // ---- fragments: contiguous k = 16*(lane/16) + s for both A and B ----
    FragAB af[4], bfr[2];
    const int ml  = lane & 15;
    const int akb = (lane < 16) ? 0 : 16;
    const int bkb = (lane < 16) ? 0 : 16;
#pragma unroll
    for (int mi = 0; mi < 4; ++mi) {
      const unsigned short* row = &As[wm + mi * 16 + ml][0];
#pragma unroll
      for (int r = 0; r < 8; ++r) {
        int kk = akb + 2 * r;
        af[mi].u[r] = *(const unsigned*)(row + kk);
      }
    }
#pragma unroll
    for (int ni = 0; ni < 2; ++ni) {
      const unsigned short* row = &Bs[wn + ni * 16 + ml][0];
#pragma unroll
      for (int r = 0; r < 8; ++r)
        bfr[ni].u[r] = *(const unsigned*)(row + bkb + 2 * r);
    }
#pragma unroll
    for (int mi = 0; mi < 4; ++mi)
#pragma unroll
      for (int ni = 0; ni < 2; ++ni)
        acc[mi][ni] = __builtin_amdgcn_wmma_f32_16x16x32_bf16(
            false, af[mi].v, false, bfr[ni].v, (short)0, acc[mi][ni], false, false);
    __syncthreads();
    if (more) {
      ldsStore();
      __syncthreads();
    }
  }

  // epilogue + store; C layout: lanes 0-15 -> M=r, lanes 16-31 -> M=8+r, N=lane%16
  const int ml = lane & 15;
#pragma unroll
  for (int mi = 0; mi < 4; ++mi)
#pragma unroll
    for (int ni = 0; ni < 2; ++ni)
#pragma unroll
      for (int r = 0; r < 8; ++r) {
        int m = m0 + wm + mi * 16 + ((lane < 16) ? r : r + 8);
        int n = n0 + wn + ni * 16 + ml;
        if (!GUARD || n < N) {
          float val = acc[mi][ni][r];
          if (EPI == 1) val = fmaxf(val, 0.0f);
          else if (EPI == 2) val += EA[(long long)m * ldc + n];
          else if (EPI == 3) {
            val += EA[(long long)hidx * S * S + (long long)m * S + n];
            val += (1.0f - (float)EB[(long long)bidx * S + n]) * -1e9f;
          }
          C[(long long)m * ldc + n] = val;
        }
      }
}

__global__ __launch_bounds__(256)
void embed_kernel(const int* __restrict__ ids, const float* __restrict__ emb,
                  float* __restrict__ x, int D) {
  long long t = blockIdx.x;
  const float4* src = (const float4*)(emb + (long long)ids[t] * D);
  float4* dst = (float4*)(x + t * D);
  for (int j = threadIdx.x; j < (D >> 2); j += 256) dst[j] = src[j];
}

__global__ __launch_bounds__(256)
void bias_kernel(const float* __restrict__ relb, float* __restrict__ bias, int S, int H) {
  int idx = blockIdx.x * 256 + threadIdx.x;
  if (idx >= S * S) return;
  int q = idx / S, k = idx % S;
  int rel = k - q;                     // memory - query
  int ret = (rel > 0) ? 16 : 0;
  int n = (rel < 0) ? -rel : rel;
  int val;
  if (n < 8) val = n;
  else {
    int vl = 8 + (int)(logf((float)n * 0.125f + 1e-6f) * 2.8853901f); // 8/ln(16)
    val = (vl < 15) ? vl : 15;
  }
  int bucket = ret + val;
  for (int h = 0; h < H; ++h)
    bias[(long long)h * S * S + idx] = relb[bucket * H + h];
}

__global__ __launch_bounds__(256)
void rmsnorm_kernel(const float* __restrict__ x, const float* __restrict__ w,
                    float* __restrict__ out, int D) {
  long long row = blockIdx.x;
  const float* xr = x + row * D;
  float* orow = out + row * D;
  float ss = 0.0f;
  for (int j = threadIdx.x; j < D; j += 256) { float v = xr[j]; ss = fmaf(v, v, ss); }
#pragma unroll
  for (int o = 16; o > 0; o >>= 1) ss += __shfl_xor(ss, o, 32);
  __shared__ float red[8];
  if ((threadIdx.x & 31) == 0) red[threadIdx.x >> 5] = ss;
  __syncthreads();
  float tot = 0.0f;
#pragma unroll
  for (int i = 0; i < 8; ++i) tot += red[i];
  float inv = rsqrtf(tot / (float)D + 1e-6f);
  for (int j = threadIdx.x; j < D; j += 256) orow[j] = w[j] * xr[j] * inv;
}

__global__ __launch_bounds__(256)
void softmax_kernel(float* __restrict__ p, int S) {   // S == 512, one wave per row
  int wave = threadIdx.x >> 5, lane = threadIdx.x & 31;
  long long row = (long long)blockIdx.x * 8 + wave;
  float* r = p + row * S;
  float vals[16];
  float mx = -1e30f;
#pragma unroll
  for (int i = 0; i < 16; ++i) { vals[i] = r[lane + i * 32]; mx = fmaxf(mx, vals[i]); }
#pragma unroll
  for (int o = 16; o > 0; o >>= 1) mx = fmaxf(mx, __shfl_xor(mx, o, 32));
  float sum = 0.0f;
#pragma unroll
  for (int i = 0; i < 16; ++i) { vals[i] = __expf(vals[i] - mx); sum += vals[i]; }
#pragma unroll
  for (int o = 16; o > 0; o >>= 1) sum += __shfl_xor(sum, o, 32);
  float inv = 1.0f / sum;
#pragma unroll
  for (int i = 0; i < 16; ++i) r[lane + i * 32] = vals[i] * inv;
}

__global__ __launch_bounds__(256)
void lnout_kernel(const float* __restrict__ ctx, const float* __restrict__ g,
                  const float* __restrict__ bb, const int* __restrict__ mask,
                  float* __restrict__ out, int D) {
  long long row = blockIdx.x;
  const float* xr = ctx + row * D;
  float* orow = out + row * D;
  __shared__ float red[8];
  float s = 0.0f;
  for (int j = threadIdx.x; j < D; j += 256) s += xr[j];
#pragma unroll
  for (int o = 16; o > 0; o >>= 1) s += __shfl_xor(s, o, 32);
  if ((threadIdx.x & 31) == 0) red[threadIdx.x >> 5] = s;
  __syncthreads();
  float tot = 0.0f;
#pragma unroll
  for (int i = 0; i < 8; ++i) tot += red[i];
  float mu = tot / (float)D;
  __syncthreads();
  float s2 = 0.0f;
  for (int j = threadIdx.x; j < D; j += 256) { float d = xr[j] - mu; s2 = fmaf(d, d, s2); }
#pragma unroll
  for (int o = 16; o > 0; o >>= 1) s2 += __shfl_xor(s2, o, 32);
  if ((threadIdx.x & 31) == 0) red[threadIdx.x >> 5] = s2;
  __syncthreads();
  float tot2 = 0.0f;
#pragma unroll
  for (int i = 0; i < 8; ++i) tot2 += red[i];
  float inv = rsqrtf(tot2 / (float)D + 1e-5f);
  float mf = (float)mask[row];
  for (int j = threadIdx.x; j < D; j += 256)
    orow[j] = ((xr[j] - mu) * inv * g[j] + bb[j]) * mf;
}

__global__ __launch_bounds__(256)
void maskout_kernel(const int* __restrict__ mask, float* __restrict__ out, int n) {
  int i = blockIdx.x * 256 + threadIdx.x;
  if (i < n) out[i] = (float)mask[i];
}

extern "C" void kernel_launch(void* const* d_in, const int* in_sizes, int n_in,
                              void* d_out, int out_size, void* d_ws, size_t ws_size,
                              hipStream_t stream) {
  (void)in_sizes; (void)n_in; (void)out_size; (void)ws_size;
  const int*   ids   = (const int*)d_in[0];
  const int*   amask = (const int*)d_in[1];
  const float* emb   = (const float*)d_in[2];
  const float* wq    = (const float*)d_in[3];
  const float* wk    = (const float*)d_in[4];
  const float* wv    = (const float*)d_in[5];
  const float* wo    = (const float*)d_in[6];
  const float* ln1   = (const float*)d_in[7];
  const float* wi    = (const float*)d_in[8];
  const float* wof   = (const float*)d_in[9];
  const float* ln2   = (const float*)d_in[10];
  const float* fln   = (const float*)d_in[11];
  const float* relb  = (const float*)d_in[12];
  const float* pw    = (const float*)d_in[13];
  const float* pg    = (const float*)d_in[14];
  const float* pb    = (const float*)d_in[15];
  float* out = (float*)d_out;

  const int Bn = 8, S = 512, Dm = 1024, H = 16, Dk = 64, Dff = 4096, L = 6, CTX = 4096;
  const int BS = Bn * S;

  char* wsp = (char*)d_ws;
  auto alloc = [&](size_t elems) { float* p = (float*)wsp; wsp += elems * sizeof(float); return p; };
  float* xbuf  = alloc((size_t)BS * Dm);
  float* hbuf  = alloc((size_t)BS * Dm);   // rmsnorm out; re-used for attention output
  float* qbuf  = alloc((size_t)BS * Dm);
  float* kbuf  = alloc((size_t)BS * Dm);
  float* vbuf  = alloc((size_t)BS * Dm);
  float* biasb = alloc((size_t)H * S * S);
  float* big   = alloc((size_t)Bn * H * S * S);   // aliased: scores / FF / ctx
  float* scores = big;
  float* ffbuf  = big;
  float* ctxbuf = big;

  embed_kernel<<<BS, 256, 0, stream>>>(ids, emb, xbuf, Dm);
  bias_kernel<<<(S * S + 255) / 256, 256, 0, stream>>>(relb, biasb, S, H);

  dim3 gDm((Dm + 127) / 128, (BS + 127) / 128, 1);
  dim3 gFF((Dff + 127) / 128, (BS + 127) / 128, 1);
  dim3 gSC((S + 127) / 128, (S + 127) / 128, Bn * H);
  dim3 gAV((Dk + 127) / 128, (S + 127) / 128, Bn * H);
  dim3 gPJ((CTX + 127) / 128, (BS + 127) / 128, 1);

  for (int l = 0; l < L; ++l) {
    rmsnorm_kernel<<<BS, 256, 0, stream>>>(xbuf, ln1 + (size_t)l * Dm, hbuf, Dm);
    gemm_wmma<0, false, false><<<gDm, 256, 0, stream>>>(hbuf, Dm, 0, 0,
        wq + (size_t)l * Dm * Dm, Dm, 0, 0, qbuf, Dm, 0, 0,
        BS, Dm, Dm, 1, nullptr, nullptr, S);
    gemm_wmma<0, false, false><<<gDm, 256, 0, stream>>>(hbuf, Dm, 0, 0,
        wk + (size_t)l * Dm * Dm, Dm, 0, 0, kbuf, Dm, 0, 0,
        BS, Dm, Dm, 1, nullptr, nullptr, S);
    gemm_wmma<0, false, false><<<gDm, 256, 0, stream>>>(hbuf, Dm, 0, 0,
        wv + (size_t)l * Dm * Dm, Dm, 0, 0, vbuf, Dm, 0, 0,
        BS, Dm, Dm, 1, nullptr, nullptr, S);
    // scores[b,h] = q kᵀ + bias[h] + mask[b]   (A: [S,64] ld Dm, B: kᵀ TRANSB)
    gemm_wmma<3, true, false><<<gSC, 256, 0, stream>>>(qbuf, Dm, (long long)S * Dm, Dk,
        kbuf, Dm, (long long)S * Dm, Dk,
        scores, S, (long long)H * S * S, (long long)S * S,
        S, S, Dk, H, biasb, amask, S);
    softmax_kernel<<<(Bn * H * S) / 8, 256, 0, stream>>>(scores, S);
    // o[b,h] = softmax(scores) @ v  -> hbuf[b, q, h*64+d]   (N=64 -> GUARD)
    gemm_wmma<0, false, true><<<gAV, 256, 0, stream>>>(scores, S, (long long)H * S * S, (long long)S * S,
        vbuf, Dm, (long long)S * Dm, Dk,
        hbuf, Dm, (long long)S * Dm, Dk,
        S, Dk, S, H, nullptr, nullptr, S);
    // x += o @ wo
    gemm_wmma<2, false, false><<<gDm, 256, 0, stream>>>(hbuf, Dm, 0, 0,
        wo + (size_t)l * Dm * Dm, Dm, 0, 0, xbuf, Dm, 0, 0,
        BS, Dm, Dm, 1, xbuf, nullptr, S);
    rmsnorm_kernel<<<BS, 256, 0, stream>>>(xbuf, ln2 + (size_t)l * Dm, hbuf, Dm);
    gemm_wmma<1, false, false><<<gFF, 256, 0, stream>>>(hbuf, Dm, 0, 0,
        wi + (size_t)l * Dm * Dff, Dff, 0, 0, ffbuf, Dff, 0, 0,
        BS, Dff, Dm, 1, nullptr, nullptr, S);
    gemm_wmma<2, false, false><<<gDm, 256, 0, stream>>>(ffbuf, Dff, 0, 0,
        wof + (size_t)l * Dff * Dm, Dm, 0, 0, xbuf, Dm, 0, 0,
        BS, Dm, Dff, 1, xbuf, nullptr, S);
  }

  rmsnorm_kernel<<<BS, 256, 0, stream>>>(xbuf, fln, hbuf, Dm);
  gemm_wmma<0, false, false><<<gPJ, 256, 0, stream>>>(hbuf, Dm, 0, 0,
      pw, CTX, 0, 0, ctxbuf, CTX, 0, 0, BS, CTX, Dm, 1, nullptr, nullptr, S);
  lnout_kernel<<<BS, 256, 0, stream>>>(ctxbuf, pg, pb, amask, out, CTX);
  maskout_kernel<<<(BS + 255) / 256, 256, 0, stream>>>(amask, out + (size_t)BS * CTX, BS);
}
